// Rotatory_Attention_Unet_v3_42417097015762
// MI455X (gfx1250) — matrix operand order repaired and checked by reading, not compiled, not run on hardware
//
#include <hip/hip_runtime.h>
#include <hip/hip_bf16.h>
#include <math.h>

// ---------------------------------------------------------------------------
// CDNA5 (gfx1250) attention U-Net forward.
// All dense math (3x3/1x1 convs as implicit GEMM, linear-attention GEMMs)
// runs on v_wmma_f32_16x16x32_f16 (wave32), f32 accumulate.
// 64x64 tiles, 8 waves/block, 2 WMMAs/wave/K-step, LDS fragments laid out so
// each lane reads ds_load_b128 pairs; global loads are branchless float4 with
// register fetch-ahead overlapping WMMA.
// ---------------------------------------------------------------------------

#define KC 32

typedef __attribute__((ext_vector_type(16))) _Float16 v16h;
typedef __attribute__((ext_vector_type(8)))  _Float16 v8h;
typedef __attribute__((ext_vector_type(4)))  _Float16 v4h;
typedef __attribute__((ext_vector_type(8)))  float    v8f;

__device__ __forceinline__ v16h cat16(v8h lo, v8h hi) {
  return __builtin_shufflevector(lo, hi, 0, 1, 2, 3, 4, 5, 6, 7,
                                 8, 9, 10, 11, 12, 13, 14, 15);
}

__device__ __forceinline__ float apply_epi(float v, int epi) {
  switch (epi) {
    case 1: return v > 0.f ? v : 0.f;                 // relu
    case 2: return v > 0.f ? v + 1.f : __expf(v);     // elu(v)+1
    case 3: return 1.f / (1.f + __expf(-v));          // sigmoid
    default: return v;
  }
}

// ---------------------------------------------------------------------------
// Generic batched GEMM: C[M,N] = epi(A[M,K] * B[K,N] + bias[N])
// transA: logical A[m][k] = Aptr[k*lda + m]
// Assumes (verified at every call site): K % 32 == 0, N % 64 == 0,
// lda/ldb % 4 == 0; M arbitrary for !transA, M % 4 == 0 for transA.
// ---------------------------------------------------------------------------
__global__ void gemm_wmma_kernel(const float* __restrict__ A,
                                 const float* __restrict__ Bm,
                                 const float* __restrict__ bias,
                                 float* __restrict__ C,
                                 int M, int N, int K, int lda, int ldb, int ldc,
                                 long long sA, long long sB, long long sC,
                                 int transA, int epi) {
  __shared__ __align__(16) _Float16 Ash[64][40];  // [M][K], 80B row stride
  __shared__ __align__(16) _Float16 Bsh[64][40];  // [N][K] (transposed)
  const int tid  = threadIdx.x;
  const int lane = tid & 31;
  const int wave = tid >> 5;
  const int waveM = wave >> 1, waveN = wave & 1;
  const int tileM = blockIdx.y * 64, tileN = blockIdx.x * 64;
  const long long z = blockIdx.z;
  const float* Ab = A + z * sA;
  const float* Bb = Bm + z * sB;
  float* Cb = C + z * sC;

  // A no-trans: thread j<2 loads float4 at (row arj0+32j, cols ac4..ac4+3)
  const int arj0 = tid >> 3;
  const int ac4  = (tid & 7) * 4;
  const float* arow[2]; float amask[2] = {1.f, 1.f};
  // A trans: thread j<2 loads float4 along M at (k-col tac, rows tar0+32j..+3)
  const int tac  = tid & 31;
  const int tar0 = (tid >> 5) * 4;
  const float* tcol[2] = {Ab, Ab}; float tmask[2] = {1.f, 1.f};
  int toff[2] = {0, 0};
  if (!transA) {
    for (int j = 0; j < 2; ++j) {
      int m = tileM + arj0 + 32 * j;
      bool ok = m < M;
      arow[j] = Ab + (long long)(ok ? m : 0) * lda;
      amask[j] = ok ? 1.f : 0.f;
    }
  } else {
    for (int j = 0; j < 2; ++j) {
      int m = tileM + tar0 + 32 * j;
      bool ok = m < M;  // M % 4 == 0 => whole float4 valid together
      toff[j] = ok ? m : 0;
      tmask[j] = ok ? 1.f : 0.f;
    }
  }
  // B: thread j<2 loads float4 at (k-row brj0+16j, n-cols bc4..+3)
  const int brj0 = tid >> 4;
  const int bc4  = (tid & 15) * 4;

  float4 fa[2], fb[2];
  auto fetch = [&](int k0) {
    if (!transA) {
      fa[0] = *(const float4*)(arow[0] + k0 + ac4);
      fa[1] = *(const float4*)(arow[1] + k0 + ac4);
    } else {
      fa[0] = *(const float4*)(Ab + (long long)(k0 + tac) * lda + toff[0]);
      fa[1] = *(const float4*)(Ab + (long long)(k0 + tac) * lda + toff[1]);
    }
    fb[0] = *(const float4*)(Bb + (long long)(k0 + brj0) * ldb + tileN + bc4);
    fb[1] = *(const float4*)(Bb + (long long)(k0 + brj0 + 16) * ldb + tileN + bc4);
  };
  auto stage = [&]() {
    if (!transA) {
#pragma unroll
      for (int j = 0; j < 2; ++j) {
        v4h h;
        h[0] = (_Float16)(fa[j].x * amask[j]);
        h[1] = (_Float16)(fa[j].y * amask[j]);
        h[2] = (_Float16)(fa[j].z * amask[j]);
        h[3] = (_Float16)(fa[j].w * amask[j]);
        *(v4h*)&Ash[arj0 + 32 * j][ac4] = h;  // 8B-aligned ds_store_b64
      }
    } else {
#pragma unroll
      for (int j = 0; j < 2; ++j) {
        Ash[tar0 + 32 * j + 0][tac] = (_Float16)(fa[j].x * tmask[j]);
        Ash[tar0 + 32 * j + 1][tac] = (_Float16)(fa[j].y * tmask[j]);
        Ash[tar0 + 32 * j + 2][tac] = (_Float16)(fa[j].z * tmask[j]);
        Ash[tar0 + 32 * j + 3][tac] = (_Float16)(fa[j].w * tmask[j]);
      }
    }
#pragma unroll
    for (int j = 0; j < 2; ++j) {
      int r = brj0 + 16 * j;
      Bsh[bc4 + 0][r] = (_Float16)fb[j].x;
      Bsh[bc4 + 1][r] = (_Float16)fb[j].y;
      Bsh[bc4 + 2][r] = (_Float16)fb[j].z;
      Bsh[bc4 + 3][r] = (_Float16)fb[j].w;
    }
  };

  v8f acc0 = {}, acc1 = {};
  const int rr = lane & 15;
  const int kA = (lane >= 16) ? 8 : 0;
  const int kB = (lane >= 16) ? 8 : 0;

  fetch(0);
  for (int k0 = 0; k0 < K; k0 += KC) {
    stage();
    __syncthreads();
    if (k0 + KC < K) fetch(k0 + KC);  // overlap next slab's loads with WMMA
    v8h alo = *(const v8h*)&Ash[waveM * 16 + rr][kA];
    v8h ahi = *(const v8h*)&Ash[waveM * 16 + rr][16 + kA];
    v16h a = cat16(alo, ahi);
    const int n0 = waveN * 32 + rr;
    v16h b0 = cat16(*(const v8h*)&Bsh[n0][kB], *(const v8h*)&Bsh[n0][kB + 16]);
    v16h b1 = cat16(*(const v8h*)&Bsh[n0 + 16][kB], *(const v8h*)&Bsh[n0 + 16][kB + 16]);
    acc0 = __builtin_amdgcn_wmma_f32_16x16x32_f16(false, a, false, b0, (short)0, acc0, false, false);
    acc1 = __builtin_amdgcn_wmma_f32_16x16x32_f16(false, a, false, b1, (short)0, acc1, false, false);
    __syncthreads();
  }
  const bool hi = lane >= 16;
#pragma unroll
  for (int sub = 0; sub < 2; ++sub) {
    int n = tileN + waveN * 32 + sub * 16 + rr;
    if (n < N) {
      float bb = bias ? bias[n] : 0.f;
      const v8f acc = sub ? acc1 : acc0;
#pragma unroll
      for (int i = 0; i < 8; ++i) {
        int m = tileM + waveM * 16 + i + (hi ? 8 : 0);
        if (m < M) Cb[(long long)m * ldc + n] = apply_epi(acc[i] + bb, epi);
      }
    }
  }
}

// ---------------------------------------------------------------------------
// Implicit-GEMM conv (SAME, stride 1), dual-source channel concat:
//   channels [0,c0) from S0, [c0,c0+c1) from S1.  NCHW f32 in/out.
// M = Bn*H*W is a multiple of 64 for every layer here.
// ---------------------------------------------------------------------------
template <int KSZ>
__global__ void conv_wmma_kernel(const float* __restrict__ S0, int c0,
                                 const float* __restrict__ S1, int c1,
                                 const float* __restrict__ Wt,
                                 const float* __restrict__ bias,
                                 float* __restrict__ Out,
                                 int Bn, int H, int Wd, int Cout, int epi) {
  constexpr int KK2 = KSZ * KSZ;
  constexpr int PAD = KSZ / 2;
  const int Cin = c0 + c1;
  const int K = Cin * KK2;
  const int HW = H * Wd;
  const int M = Bn * HW;
  __shared__ __align__(16) _Float16 Ash[64][40];
  __shared__ __align__(16) _Float16 Bsh[64][40];  // [n][k] transposed
  const int tid  = threadIdx.x;
  const int lane = tid & 31;
  const int wave = tid >> 5;
  const int waveM = wave >> 1, waveN = wave & 1;
  const int tileM = blockIdx.y * 64, tileN = blockIdx.x * 64;
  const float* S1s = S1 ? S1 : S0;

  // A gather mapping: fixed k-column cA per thread, rows rA0+8j (j<8).
  // Row decode hoisted out of the K loop.
  const int cA  = tid & 31;
  const int rA0 = tid >> 5;
  const float* base0[8]; const float* base1[8];
  int ay[8], ax[8];
#pragma unroll
  for (int j = 0; j < 8; ++j) {
    int p = tileM + rA0 + 8 * j;
    int b = p / HW, rem = p - b * HW;
    ay[j] = rem / Wd; ax[j] = rem - ay[j] * Wd;
    base0[j] = S0 + (long long)b * c0 * HW;
    base1[j] = S1s + (long long)b * c1 * HW;
  }
  // Weight mapping: same k-column cA, n rows nW0+8j.
  const int nW0 = tid >> 5;
  const float* wrow[8]; float wmask[8];
#pragma unroll
  for (int j = 0; j < 8; ++j) {
    int n = tileN + nW0 + 8 * j;
    bool ok = n < Cout;
    wrow[j] = Wt + (long long)(ok ? n : 0) * K;
    wmask[j] = ok ? 1.f : 0.f;
  }

  float fa[8], fw[8];
  auto fetch = [&](int k0) {
    int k = k0 + cA;
    bool kok = k < K;
    int kk = kok ? k : 0;
    int ci = kk / KK2;                       // compile-time KK2 => cheap
    int off = kk - ci * KK2;
    int ky = off / KSZ - PAD;
    int kx = off - (off / KSZ) * KSZ - PAD;
    bool losrc = ci < c0;
    long long choff = (long long)(losrc ? ci : ci - c0) * HW;
#pragma unroll
    for (int j = 0; j < 8; ++j) {
      int yy = ay[j] + ky, xx = ax[j] + kx;
      bool ok = kok && (unsigned)yy < (unsigned)H && (unsigned)xx < (unsigned)Wd;
      const float* p = (losrc ? base0[j] : base1[j]) + choff + (long long)yy * Wd + xx;
      const float* ps = ok ? p : S0;  // branchless clamp
      float v = *ps;
      fa[j] = ok ? v : 0.f;
    }
#pragma unroll
    for (int j = 0; j < 8; ++j) {
      float v = wrow[j][kk];
      fw[j] = (kok ? v : 0.f) * wmask[j];
    }
  };
  auto stage = [&]() {
#pragma unroll
    for (int j = 0; j < 8; ++j) Ash[rA0 + 8 * j][cA] = (_Float16)fa[j];
#pragma unroll
    for (int j = 0; j < 8; ++j) Bsh[nW0 + 8 * j][cA] = (_Float16)fw[j];
  };

  v8f acc0 = {}, acc1 = {};
  const int rr = lane & 15;
  const int kA = (lane >= 16) ? 8 : 0;
  const int kB = (lane >= 16) ? 8 : 0;

  fetch(0);
  for (int k0 = 0; k0 < K; k0 += KC) {
    stage();
    __syncthreads();
    if (k0 + KC < K) fetch(k0 + KC);
    v8h alo = *(const v8h*)&Ash[waveM * 16 + rr][kA];
    v8h ahi = *(const v8h*)&Ash[waveM * 16 + rr][16 + kA];
    v16h a = cat16(alo, ahi);
    const int n0 = waveN * 32 + rr;
    v16h b0 = cat16(*(const v8h*)&Bsh[n0][kB], *(const v8h*)&Bsh[n0][kB + 16]);
    v16h b1 = cat16(*(const v8h*)&Bsh[n0 + 16][kB], *(const v8h*)&Bsh[n0 + 16][kB + 16]);
    acc0 = __builtin_amdgcn_wmma_f32_16x16x32_f16(false, a, false, b0, (short)0, acc0, false, false);
    acc1 = __builtin_amdgcn_wmma_f32_16x16x32_f16(false, a, false, b1, (short)0, acc1, false, false);
    __syncthreads();
  }
  const bool hi = lane >= 16;
  int mb[8], my[8], mx[8];
#pragma unroll
  for (int i = 0; i < 8; ++i) {
    int m = tileM + waveM * 16 + i + (hi ? 8 : 0);
    mb[i] = m / HW;
    int rem = m - mb[i] * HW;
    my[i] = rem / Wd; mx[i] = rem - my[i] * Wd;
  }
#pragma unroll
  for (int sub = 0; sub < 2; ++sub) {
    int n = tileN + waveN * 32 + sub * 16 + rr;
    if (n < Cout) {
      float bb = bias ? bias[n] : 0.f;
      const v8f acc = sub ? acc1 : acc0;
#pragma unroll
      for (int i = 0; i < 8; ++i) {
        Out[(((long long)mb[i] * Cout + n) * H + my[i]) * Wd + mx[i]] =
            apply_epi(acc[i] + bb, epi);
      }
    }
  }
}

// ---------------------------------------------------------------------------
// BatchNorm: per-channel mean/var over (N,H,W), then apply (+optional relu)
// ---------------------------------------------------------------------------
__global__ void bn_stats_kernel(const float* __restrict__ X, int Bn, int C, int HW,
                                float* __restrict__ mean, float* __restrict__ rstd) {
  const int c = blockIdx.x;
  const int tid = threadIdx.x;
  __shared__ float sh1[256], sh2[256];
  float sum = 0.f, sq = 0.f;
  const long long per = (long long)Bn * HW;
  for (long long i = tid; i < per; i += 256) {
    long long b = i / HW, r = i - b * HW;
    float v = X[(b * C + c) * (long long)HW + r];
    sum += v; sq += v * v;
  }
  sh1[tid] = sum; sh2[tid] = sq;
  __syncthreads();
  for (int s = 128; s > 0; s >>= 1) {
    if (tid < s) { sh1[tid] += sh1[tid + s]; sh2[tid] += sh2[tid + s]; }
    __syncthreads();
  }
  if (tid == 0) {
    float m = sh1[0] / (float)per;
    float var = sh2[0] / (float)per - m * m;
    mean[c] = m;
    rstd[c] = rsqrtf(var + 1e-5f);
  }
}

__global__ void bn_apply_kernel(float* __restrict__ X, const float* __restrict__ mean,
                                const float* __restrict__ rstd,
                                const float* __restrict__ gamma,
                                const float* __restrict__ beta,
                                int C, int HW, long long total, int relu) {
  long long i = (long long)blockIdx.x * 256 + threadIdx.x;
  if (i >= total) return;
  int c = (int)((i / HW) % C);
  float v = (X[i] - mean[c]) * rstd[c] * gamma[c] + beta[c];
  if (relu) v = v > 0.f ? v : 0.f;
  X[i] = v;
}

// ---------------------------------------------------------------------------
// Elementwise helpers
// ---------------------------------------------------------------------------
__global__ void maxpool_kernel(const float* __restrict__ X, float* __restrict__ Y,
                               int BC, int H, int W) {
  const int OH = H >> 1, OW = W >> 1;
  long long i = (long long)blockIdx.x * 256 + threadIdx.x;
  long long tot = (long long)BC * OH * OW;
  if (i >= tot) return;
  int bc = (int)(i / (OH * OW));
  int rem = (int)(i - (long long)bc * OH * OW);
  int oy = rem / OW, ox = rem - oy * OW;
  const float* p = X + (long long)bc * H * W;
  float a = p[(2 * oy) * W + 2 * ox],     b = p[(2 * oy) * W + 2 * ox + 1];
  float c = p[(2 * oy + 1) * W + 2 * ox], d = p[(2 * oy + 1) * W + 2 * ox + 1];
  Y[i] = fmaxf(fmaxf(a, b), fmaxf(c, d));
}

// upsample2x with the reference's linspace(0, H-1, 2H) sampling grid
__global__ void upsample_kernel(const float* __restrict__ X, float* __restrict__ Y,
                                int BC, int H, int W) {
  const int OH = 2 * H, OW = 2 * W;
  long long i = (long long)blockIdx.x * 256 + threadIdx.x;
  long long tot = (long long)BC * OH * OW;
  if (i >= tot) return;
  int bc = (int)(i / (OH * OW));
  int rem = (int)(i - (long long)bc * OH * OW);
  int oy = rem / OW, ox = rem - oy * OW;
  float sy = (float)(H - 1) / (float)(OH - 1);
  float sx = (float)(W - 1) / (float)(OW - 1);
  float ys = oy * sy, xs = ox * sx;
  int y0 = (int)floorf(ys), x0 = (int)floorf(xs);
  float fy = ys - y0, fx = xs - x0;
  int y1 = min(y0 + 1, H - 1), x1 = min(x0 + 1, W - 1);
  const float* p = X + (long long)bc * H * W;
  float r0 = p[y0 * W + x0] * (1.f - fx) + p[y0 * W + x1] * fx;
  float r1 = p[y1 * W + x0] * (1.f - fx) + p[y1 * W + x1] * fx;
  Y[i] = r0 * (1.f - fy) + r1 * fy;
}

__global__ void add_relu_kernel(float* __restrict__ A, const float* __restrict__ B,
                                long long total) {
  long long i = (long long)blockIdx.x * 256 + threadIdx.x;
  if (i >= total) return;
  float v = A[i] + B[i];
  A[i] = v > 0.f ? v : 0.f;
}

__global__ void mul_kernel(float* __restrict__ A, const float* __restrict__ B,
                           long long total) {
  long long i = (long long)blockIdx.x * 256 + threadIdx.x;
  if (i >= total) return;
  A[i] *= B[i];
}

// seq[(b*N+n)*C+c] = X[(b*C+c)*N+n]   (NCHW bottleneck -> (B, N, C) rows)
__global__ void transpose_seq_kernel(const float* __restrict__ X, float* __restrict__ S,
                                     int Bn, int C, int N) {
  long long i = (long long)blockIdx.x * 256 + threadIdx.x;
  long long tot = (long long)Bn * C * N;
  if (i >= tot) return;
  int b = (int)(i / ((long long)C * N));
  int rem = (int)(i - (long long)b * C * N);
  int n = rem / C, c = rem - n * C;
  S[i] = X[((long long)b * C + c) * N + n];
}

// Ksum[t][h] = sum_n K[(t*256+n)*768 + h]
__global__ void ksum_kernel(const float* __restrict__ K, float* __restrict__ Ksum) {
  int i = blockIdx.x * 256 + threadIdx.x;
  if (i >= 14 * 768) return;
  int t = i / 768, h = i - t * 768;
  float s = 0.f;
  const float* p = K + (long long)t * 256 * 768 + h;
  for (int n = 0; n < 256; ++n) s += p[(long long)n * 768];
  Ksum[i] = s;
}

// Z[t][n] = dot(Q[t][n][:], Ksum[t][:])
__global__ void zdot_kernel(const float* __restrict__ Q, const float* __restrict__ Ksum,
                            float* __restrict__ Z) {
  int t = blockIdx.x, n = threadIdx.x;  // 14 blocks x 256 threads
  const float* q = Q + ((long long)t * 256 + n) * 768;
  const float* k = Ksum + (long long)t * 768;
  float s = 0.f;
  for (int h = 0; h < 768; ++h) s += q[h] * k[h];
  Z[t * 256 + n] = s;
}

// cat[row][br*64+d] = O[row][d] / (Z[row] + 1e-6)
__global__ void scale_cat_kernel(const float* __restrict__ O, const float* __restrict__ Z,
                                 float* __restrict__ cat, int br) {
  long long i = (long long)blockIdx.x * 256 + threadIdx.x;
  if (i >= (long long)14 * 256 * 64) return;
  int row = (int)(i >> 6), d = (int)(i & 63);
  cat[(long long)row * 192 + br * 64 + d] = O[i] / (Z[row] + 1e-6f);
}

// per row (t,n): softmax over C of R, multiply into X image t+1 in place
__global__ void softmax_mul_kernel(const float* __restrict__ R, float* __restrict__ X,
                                   int C, int Nn) {
  const int row = blockIdx.x;
  const int t = row / Nn, n = row - t * Nn;
  const int tid = threadIdx.x;  // 128 threads, 4 channels each (C=512)
  __shared__ float sh[128];
  float vals[4];
  float lmax = -1e30f;
#pragma unroll
  for (int j = 0; j < 4; ++j) {
    int c = tid + j * 128;
    vals[j] = R[(long long)row * C + c];
    lmax = fmaxf(lmax, vals[j]);
  }
  sh[tid] = lmax; __syncthreads();
  for (int s = 64; s > 0; s >>= 1) {
    if (tid < s) sh[tid] = fmaxf(sh[tid], sh[tid + s]);
    __syncthreads();
  }
  float mx = sh[0]; __syncthreads();
  float lsum = 0.f;
#pragma unroll
  for (int j = 0; j < 4; ++j) { vals[j] = __expf(vals[j] - mx); lsum += vals[j]; }
  sh[tid] = lsum; __syncthreads();
  for (int s = 64; s > 0; s >>= 1) {
    if (tid < s) sh[tid] += sh[tid + s];
    __syncthreads();
  }
  float inv = 1.f / sh[0];
#pragma unroll
  for (int j = 0; j < 4; ++j) {
    int c = tid + j * 128;
    long long xi = (((long long)(t + 1) * C + c) * Nn) + n;
    X[xi] = X[xi] * (vals[j] * inv);
  }
}

// ---------------------------------------------------------------------------
// Host orchestration
// ---------------------------------------------------------------------------
extern "C" void kernel_launch(void* const* d_in, const int* in_sizes, int n_in,
                              void* d_out, int out_size, void* d_ws, size_t ws_size,
                              hipStream_t stream) {
  (void)in_sizes; (void)n_in; (void)out_size; (void)ws_size;
  const float* x = (const float*)d_in[0];
  auto P = [&](int i) { return (const float*)d_in[i]; };
  // Flattened param order (dict insertion order, leaves recursively):
  // 0:x | e1 w1 b2 g3 be4 | e2 5..8 | e3 9..12 | b1 13..16
  // rag: 17 Wq_l 18 Wk_l 19 Wv_l 20 Wq_t 21 Wk_t 22 Wv_t 23 Wq_r 24 Wk_r 25 Wv_r 26 Wo
  // d1: ag 27..36 (wg_w,wg_b,wg_g,wg_be, ws_w,ws_b,ws_g,ws_be, o_w,o_b), c1 37..40
  // d2: ag 41..50, c1 51..54 | d3: ag 55..64, c1 65..68 | 69 out_w 70 out_b

  const int B = 16;
  float* wsf = (float*)d_ws;
  size_t off = 0;
  auto alloc = [&](size_t n) { float* p = wsf + off; off += n; return p; };

  float* s1    = alloc((size_t)16 * 64 * 128 * 128);
  float* poolb = alloc((size_t)16 * 64 * 64 * 64);   // reused p1/p2/p3
  float* s2    = alloc((size_t)16 * 128 * 64 * 64);
  float* s3    = alloc((size_t)16 * 256 * 32 * 32);
  float* b1    = alloc((size_t)16 * 512 * 16 * 16);
  float* seq   = alloc((size_t)16 * 256 * 512);
  float* Qb    = alloc((size_t)14 * 256 * 768);
  float* Kb    = alloc((size_t)14 * 256 * 768);
  float* Vb    = alloc((size_t)14 * 256 * 64);
  float* KVb   = alloc((size_t)14 * 768 * 64);
  float* Ksum  = alloc((size_t)14 * 768);
  float* Zb    = alloc((size_t)14 * 256);
  float* tmpO  = alloc((size_t)14 * 256 * 64);
  float* catb  = alloc((size_t)14 * 256 * 192);
  float* rago  = alloc((size_t)14 * 256 * 512);
  float* meanb = alloc(512);
  float* rstdb = alloc(512);
  float* upb   = alloc((size_t)16 * 128 * 128 * 128); // max: d3 upsample
  float* wgb   = alloc((size_t)16 * 64 * 128 * 128);
  float* wsb   = alloc((size_t)16 * 64 * 128 * 128);
  float* dec1  = alloc((size_t)16 * 256 * 32 * 32);
  float* dec2  = alloc((size_t)16 * 128 * 64 * 64);
  float* dec3  = alloc((size_t)16 * 64 * 128 * 128);

  auto conv = [&](const float* s0, int c0, const float* sx, int c1, const float* w,
                  const float* bias, float* outp, int H, int W, int Cout, int ksz, int epi) {
    int M = B * H * W;
    dim3 g((Cout + 63) / 64, (M + 63) / 64, 1);
    if (ksz == 3)
      conv_wmma_kernel<3><<<g, dim3(256), 0, stream>>>(s0, c0, sx, c1, w, bias, outp,
                                                       B, H, W, Cout, epi);
    else
      conv_wmma_kernel<1><<<g, dim3(256), 0, stream>>>(s0, c0, sx, c1, w, bias, outp,
                                                       B, H, W, Cout, epi);
  };
  auto bn = [&](float* t, int C, int H, int W, const float* g, const float* be, int relu) {
    bn_stats_kernel<<<dim3(C), dim3(256), 0, stream>>>(t, B, C, H * W, meanb, rstdb);
    long long tot = (long long)B * C * H * W;
    bn_apply_kernel<<<dim3((unsigned)((tot + 255) / 256)), dim3(256), 0, stream>>>(
        t, meanb, rstdb, g, be, C, H * W, tot, relu);
  };
  auto gemm = [&](const float* A, const float* Bm, const float* bias, float* C,
                  int M, int N, int K, int lda, int ldb, int ldc, int transA, int epi,
                  long long sA, long long sB, long long sC, int batch) {
    dim3 g((N + 63) / 64, (M + 63) / 64, batch);
    gemm_wmma_kernel<<<g, dim3(256), 0, stream>>>(A, Bm, bias, C, M, N, K, lda, ldb,
                                                  ldc, sA, sB, sC, transA, epi);
  };
  auto ew = [&](long long tot) { return dim3((unsigned)((tot + 255) / 256)); };

  // ---------------- encoder ----------------
  conv(x, 1, nullptr, 0, P(1), P(2), s1, 128, 128, 64, 3, 0);
  bn(s1, 64, 128, 128, P(3), P(4), 1);
  maxpool_kernel<<<ew((long long)B * 64 * 64 * 64), 256, 0, stream>>>(s1, poolb, B * 64, 128, 128);
  conv(poolb, 64, nullptr, 0, P(5), P(6), s2, 64, 64, 128, 3, 0);
  bn(s2, 128, 64, 64, P(7), P(8), 1);
  maxpool_kernel<<<ew((long long)B * 128 * 32 * 32), 256, 0, stream>>>(s2, poolb, B * 128, 64, 64);
  conv(poolb, 128, nullptr, 0, P(9), P(10), s3, 32, 32, 256, 3, 0);
  bn(s3, 256, 32, 32, P(11), P(12), 1);
  maxpool_kernel<<<ew((long long)B * 256 * 16 * 16), 256, 0, stream>>>(s3, poolb, B * 256, 32, 32);
  conv(poolb, 256, nullptr, 0, P(13), P(14), b1, 16, 16, 512, 3, 0);
  bn(b1, 512, 16, 16, P(15), P(16), 1);

  // ---------------- rotatory linear attention on bottleneck ----------------
  transpose_seq_kernel<<<ew((long long)16 * 512 * 256), 256, 0, stream>>>(b1, seq, 16, 512, 256);
  for (int br = 0; br < 3; ++br) {  // 0:left(kv=0..13) 1:top(1..14) 2:right(2..15)
    const float* Wq = P(17 + br * 3);
    const float* Wk = P(18 + br * 3);
    const float* Wv = P(19 + br * 3);
    const float* kvBase = seq + (long long)br * 256 * 512;
    const float* qBase  = seq + (long long)1 * 256 * 512;
    gemm(qBase,  Wq, nullptr, Qb, 14 * 256, 768, 512, 512, 768, 768, 0, 2, 0, 0, 0, 1);
    gemm(kvBase, Wk, nullptr, Kb, 14 * 256, 768, 512, 512, 768, 768, 0, 2, 0, 0, 0, 1);
    gemm(kvBase, Wv, nullptr, Vb, 14 * 256,  64, 512, 512,  64,  64, 0, 0, 0, 0, 0, 1);
    ksum_kernel<<<dim3((14 * 768 + 255) / 256), 256, 0, stream>>>(Kb, Ksum);
    // KV[t] = K[t]^T (768x256) @ V[t] (256x64)
    gemm(Kb, Vb, nullptr, KVb, 768, 64, 256, 768, 64, 64, 1, 0,
         (long long)256 * 768, (long long)256 * 64, (long long)768 * 64, 14);
    zdot_kernel<<<dim3(14), 256, 0, stream>>>(Qb, Ksum, Zb);
    // O[t] = Q[t] (256x768) @ KV[t] (768x64)
    gemm(Qb, KVb, nullptr, tmpO, 256, 64, 768, 768, 64, 64, 0, 0,
         (long long)256 * 768, (long long)768 * 64, (long long)256 * 64, 14);
    scale_cat_kernel<<<ew((long long)14 * 256 * 64), 256, 0, stream>>>(tmpO, Zb, catb, br);
  }
  gemm(catb, P(26), nullptr, rago, 14 * 256, 512, 192, 192, 512, 512, 0, 0, 0, 0, 0, 1);
  softmax_mul_kernel<<<dim3(14 * 256), 128, 0, stream>>>(rago, b1, 512, 256);

  // ---------------- decoder ----------------
  auto decoder = [&](const float* xin, int Cx, int Hin, const float* skip, int Cs,
                     float* dout, int agBase, int c1Base) {
    const int H = Hin * 2;
    const int outc = Cs;
    long long upTot = (long long)B * Cx * H * H;
    upsample_kernel<<<ew(upTot), 256, 0, stream>>>(xin, upb, B * Cx, Hin, Hin);
    // wg = BN(conv1x1(up))
    conv(upb, Cx, nullptr, 0, P(agBase + 0), P(agBase + 1), wgb, H, H, outc, 1, 0);
    bn(wgb, outc, H, H, P(agBase + 2), P(agBase + 3), 0);
    // ws = BN(conv1x1(skip))
    conv(skip, Cs, nullptr, 0, P(agBase + 4), P(agBase + 5), wsb, H, H, outc, 1, 0);
    bn(wsb, outc, H, H, P(agBase + 6), P(agBase + 7), 0);
    long long tot = (long long)B * outc * H * H;
    add_relu_kernel<<<ew(tot), 256, 0, stream>>>(wgb, wsb, tot);
    // a = sigmoid(conv1x1(relu(wg+ws)))   -> wsb
    conv(wgb, outc, nullptr, 0, P(agBase + 8), P(agBase + 9), wsb, H, H, outc, 1, 3);
    mul_kernel<<<ew(tot), 256, 0, stream>>>(wsb, skip, tot);  // a *= skip
    // conv_block(concat([up, a*skip])) with implicit channel concat
    conv(upb, Cx, wsb, outc, P(c1Base + 0), P(c1Base + 1), dout, H, H, outc, 3, 0);
    bn(dout, outc, H, H, P(c1Base + 2), P(c1Base + 3), 1);
  };
  decoder(b1,   512, 16, s3, 256, dec1, 27, 37);
  decoder(dec1, 256, 32, s2, 128, dec2, 41, 51);
  decoder(dec2, 128, 64, s1,  64, dec3, 55, 65);

  // final 1x1 conv 64 -> 8 classes
  conv(dec3, 64, nullptr, 0, P(69), P(70), (float*)d_out, 128, 128, 8, 1, 0);
}
